// NormLayer_3564822856140
// MI455X (gfx1250) — matrix operand order repaired and checked by reading, not compiled, not run on hardware
//
#include <hip/hip_runtime.h>

#define NROWS 10000
#define DDIM  512
#define BM    32
#define BN    128

typedef __attribute__((ext_vector_type(16))) __bf16       v16bf;
typedef __attribute__((ext_vector_type(8)))  float        v8f;
typedef __attribute__((ext_vector_type(4)))  unsigned int v4u;

union Frag16 { v16bf bf; v4u u[2]; };

__device__ __forceinline__ unsigned short bf16rne(float f) {
    unsigned int u = __builtin_bit_cast(unsigned int, f);
    u += 0x7FFFu + ((u >> 16) & 1u);
    return (unsigned short)(u >> 16);
}

// ---------------------------------------------------------------------------
// Kernel 1: row L2-normalize x -> norm_x (bf16). One block (128 thr) per row.
// ---------------------------------------------------------------------------
__global__ __launch_bounds__(128) void rownorm_kernel(const float* __restrict__ x,
                                                      unsigned short* __restrict__ normx) {
    const int i = blockIdx.x;
    const int t = threadIdx.x;
    const float4* xr = (const float4*)(x + (size_t)i * DDIM);
    float4 v = xr[t];                         // 128 threads * 4 = 512 elements
    float ss = v.x * v.x + v.y * v.y + v.z * v.z + v.w * v.w;
#pragma unroll
    for (int off = 16; off > 0; off >>= 1) ss += __shfl_down(ss, off);
    __shared__ float wsum[4];
    if ((t & 31) == 0) wsum[t >> 5] = ss;
    __syncthreads();
    float tot = wsum[0] + wsum[1] + wsum[2] + wsum[3];
    float sc = 1.0f / fmaxf(sqrtf(tot), 1e-12f);
    unsigned int lo = (unsigned int)bf16rne(v.x * sc) | ((unsigned int)bf16rne(v.y * sc) << 16);
    unsigned int hi = (unsigned int)bf16rne(v.z * sc) | ((unsigned int)bf16rne(v.w * sc) << 16);
    uint2 pk; pk.x = lo; pk.y = hi;
    *(uint2*)(normx + (size_t)i * DDIM + t * 4) = pk;
}

// ---------------------------------------------------------------------------
// Kernel 2: tiled transpose-convert x (f32, [N][D]) -> xT (bf16, [D][N])
// ---------------------------------------------------------------------------
__global__ __launch_bounds__(256) void transpose_kernel(const float* __restrict__ x,
                                                        unsigned short* __restrict__ xT) {
    __shared__ unsigned short tile[32][33];
    const int i0 = blockIdx.x * 32;
    const int d0 = blockIdx.y * 32;
    const int tx = threadIdx.x & 31;
    const int ty = threadIdx.x >> 5;          // 0..7
#pragma unroll
    for (int k = 0; k < 4; ++k) {
        int i = i0 + ty + k * 8;
        if (i < NROWS) tile[ty + k * 8][tx] = bf16rne(x[(size_t)i * DDIM + d0 + tx]);
    }
    __syncthreads();
#pragma unroll
    for (int k = 0; k < 4; ++k) {
        int d = d0 + ty + k * 8;
        int i = i0 + tx;
        if (i < NROWS) xT[(size_t)d * NROWS + i] = tile[tx][ty + k * 8];
    }
}

// ---------------------------------------------------------------------------
// Main fused kernel: per block = 32 output rows (2 M-tiles), 8 waves.
//   S = norm_x[i,:32] @ norm_x[j:j+128,:].T   (WMMA bf16, K=512, B reused x2)
//   P = exp(S) masked by adj  (no max shift needed: |sim| <= 1)
//   acc += P @ x[j:j+128,:]                   (WMMA bf16 via xT, B reused x2)
//   out = 2*x - acc / rowsum
// ---------------------------------------------------------------------------
__global__ __launch_bounds__(256) void contranorm_main(
    const float* __restrict__ x, const int* __restrict__ adj,
    const unsigned short* __restrict__ normx, const unsigned short* __restrict__ xT,
    float* __restrict__ out)
{
    __shared__ unsigned short Ai[BM * 520];     // 32 rows x 512 bf16, stride 520 (bank pad)
    __shared__ unsigned short Pbuf[BM * 136];   // 32 rows x 128 bf16, stride 136 (bank pad)
    __shared__ float rowsumLDS[BM];

    const int tid  = threadIdx.x;
    const int w    = tid >> 5;                  // wave id 0..7
    const int lane = tid & 31;
    const int hi   = lane >> 4;                 // lane half
    const int n    = lane & 15;
    const int ib   = blockIdx.x * BM;

    // Stage the 32 norm_x rows of this block into LDS (A operand of GEMM1).
    {
        int m = tid >> 3, g = tid & 7;          // 8 threads per row, 64 bf16 each
        int srow = min(ib + m, NROWS - 1);      // clamp for last (partial) block
        const v4u* src = (const v4u*)(normx + (size_t)srow * DDIM + g * 64);
        v4u* dst = (v4u*)(Ai + m * 520 + g * 64);
#pragma unroll
        for (int c = 0; c < 8; ++c) dst[c] = src[c];
    }
    if (tid < BM) rowsumLDS[tid] = 0.0f;
    __syncthreads();

    v8f acc[8] = {};                            // [mt*4+t]: 32x64 f32 slice per wave
    float lacc[16];
#pragma unroll
    for (int r = 0; r < 16; ++r) lacc[r] = 0.0f;

    for (int jb = 0; jb < NROWS; jb += BN) {
        const bool active = (jb + w * 16) < NROWS;   // this wave's j subtile valid?
        float p[16];
        if (active) {
            v8f S0 = {}, S1 = {};
            const int jrow = jb + w * 16 + n;
#pragma unroll
            for (int k0 = 0; k0 < DDIM; k0 += 32) {
                // B fragment (32x16): lane = column (= j row), halves split K 0-15 / 16-31
                Frag16 b;
                const unsigned short* bp = normx + (size_t)jrow * DDIM + k0 + hi * 16;
                b.u[0] = *(const v4u*)(bp);
                b.u[1] = *(const v4u*)(bp + 8);
                // A fragments for both M-tiles (B reused by 2 WMMAs)
                Frag16 a0, a1;
                a0.u[0] = *(const v4u*)(Ai + n * 520 + k0 + hi * 16);
                a0.u[1] = *(const v4u*)(Ai + n * 520 + k0 + hi * 16 + 8);
                a1.u[0] = *(const v4u*)(Ai + (n + 16) * 520 + k0 + hi * 16);
                a1.u[1] = *(const v4u*)(Ai + (n + 16) * 520 + k0 + hi * 16 + 8);
                S0 = __builtin_amdgcn_wmma_f32_16x16x32_bf16(false, a0.bf, false, b.bf,
                                                             (short)0, S0, false, false);
                S1 = __builtin_amdgcn_wmma_f32_16x16x32_bf16(false, a1.bf, false, b.bf,
                                                             (short)0, S1, false, false);
            }
            // Mask + exp (sim in [-1,1] -> exp in [1/e, e]; no max shift needed)
            const size_t jcol = (size_t)(jb + w * 16 + n);
#pragma unroll
            for (int mt = 0; mt < 2; ++mt) {
#pragma unroll
                for (int r = 0; r < 8; ++r) {
                    int m = mt * 16 + r + 8 * hi;        // C/D layout: VGPR r -> M=r(+8 hi)
                    int arow = min(ib + m, NROWS - 1);   // clamp; OOB rows never stored
                    int am = __builtin_nontemporal_load(&adj[(size_t)arow * NROWS + jcol]);
                    float s = (mt == 0) ? S0[r] : S1[r];
                    float pv = (am != 0) ? 0.0f : __expf(s);
                    p[mt * 8 + r] = pv;
                    lacc[mt * 8 + r] += pv;
                }
            }
        } else {
#pragma unroll
            for (int r = 0; r < 16; ++r) p[r] = 0.0f;
        }
        // Publish this wave's 32x16 P tiles (bf16) for all waves.
#pragma unroll
        for (int mt = 0; mt < 2; ++mt) {
#pragma unroll
            for (int r = 0; r < 8; ++r) {
                int m = mt * 16 + r + 8 * hi;
                Pbuf[m * 136 + w * 16 + n] = bf16rne(p[mt * 8 + r]);
            }
        }
        __syncthreads();

        // GEMM2: acc(32x64, this wave's D slice) += P(32x128) @ x[jb:jb+128, Dslice]
#pragma unroll
        for (int k0 = 0; k0 < BN; k0 += 32) {
            Frag16 pa0, pa1;
            pa0.u[0] = *(const v4u*)(Pbuf + n * 136 + k0 + hi * 16);
            pa0.u[1] = *(const v4u*)(Pbuf + n * 136 + k0 + hi * 16 + 8);
            pa1.u[0] = *(const v4u*)(Pbuf + (n + 16) * 136 + k0 + hi * 16);
            pa1.u[1] = *(const v4u*)(Pbuf + (n + 16) * 136 + k0 + hi * 16 + 8);
#pragma unroll
            for (int t = 0; t < 4; ++t) {
                const int drow = w * 64 + t * 16 + n;     // lane = D column
                int col = jb + k0 + hi * 16;              // j index chunk (16-aligned)
                if (col + 16 > NROWS) col = NROWS - 16;   // clamp: P==0 there, data finite
                const unsigned short* bp = xT + (size_t)drow * NROWS + col;
                Frag16 b;
                b.u[0] = *(const v4u*)(bp);
                b.u[1] = *(const v4u*)(bp + 8);
                acc[t]     = __builtin_amdgcn_wmma_f32_16x16x32_bf16(false, pa0.bf, false, b.bf,
                                                                     (short)0, acc[t], false, false);
                acc[4 + t] = __builtin_amdgcn_wmma_f32_16x16x32_bf16(false, pa1.bf, false, b.bf,
                                                                     (short)0, acc[4 + t], false, false);
            }
        }
        __syncthreads();
    }

    // Reduce per-lane partial row sums (each (lane, mt, r) owns a unique (m, n) cell).
#pragma unroll
    for (int r = 0; r < 16; ++r) {
        int m = (r >> 3) * 16 + (r & 7) + 8 * hi;
        atomicAdd(&rowsumLDS[m], lacc[r]);
    }
    __syncthreads();

    float inv[16];
#pragma unroll
    for (int r = 0; r < 16; ++r)
        inv[r] = 1.0f / rowsumLDS[(r >> 3) * 16 + (r & 7) + 8 * hi];

    // out = (1+SCALE)*x - SCALE * acc / rowsum  (SCALE = 1)
#pragma unroll
    for (int mt = 0; mt < 2; ++mt) {
#pragma unroll
        for (int t = 0; t < 4; ++t) {
            const int d = w * 64 + t * 16 + n;
#pragma unroll
            for (int r = 0; r < 8; ++r) {
                int row = ib + mt * 16 + r + 8 * hi;
                if (row < NROWS) {
                    size_t idx = (size_t)row * DDIM + d;
                    float xv = x[idx];
                    __builtin_nontemporal_store(2.0f * xv - acc[mt * 4 + t][r] * inv[mt * 8 + r],
                                                &out[idx]);
                }
            }
        }
    }
}

// ---------------------------------------------------------------------------
extern "C" void kernel_launch(void* const* d_in, const int* in_sizes, int n_in,
                              void* d_out, int out_size, void* d_ws, size_t ws_size,
                              hipStream_t stream) {
    const float* x  = (const float*)d_in[0];
    const int* adj  = (const int*)d_in[1];
    float* out      = (float*)d_out;

    // Workspace: norm_x bf16 [N][D] then xT bf16 [D][N]  (~20.5 MB total)
    unsigned short* normx = (unsigned short*)d_ws;
    unsigned short* xT    = normx + (size_t)NROWS * DDIM;

    rownorm_kernel<<<NROWS, 128, 0, stream>>>(x, normx);
    transpose_kernel<<<dim3((NROWS + 31) / 32, DDIM / 32), 256, 0, stream>>>(x, xT);
    contranorm_main<<<(NROWS + BM - 1) / BM, 256, 0, stream>>>(x, adj, normx, xT, out);
}
